// GRU_15436112461845
// MI455X (gfx1250) — compile-verified, matrix operand order checked by reading, not run on hardware
//
#include <hip/hip_runtime.h>
#include <hip/hip_bf16.h>
#include <stdint.h>

typedef __bf16 bf16_t;
typedef __attribute__((ext_vector_type(16))) __bf16 v16bf;
typedef __attribute__((ext_vector_type(8)))  float  v8f;

#define B_N 512
#define T_N 1024
#define IN_N 64
#define H_N 128
#define G_N 384   // 3*H
#define NT_N 24   // G_N/16 N-tiles

__device__ __forceinline__ float sigmoidf_(float x) { return 1.f / (1.f + __expf(-x)); }

__device__ __forceinline__ v8f wmma_bf16(v16bf a, v16bf b, v8f c) {
  // D = A(16x32 bf16) * B(32x16 bf16) + C(16x16 f32)
  return __builtin_amdgcn_wmma_f32_16x16x32_bf16(false, a, false, b, (short)0, c, false, false);
}

// Load a B-fragment: 16 contiguous bf16 (32B) per lane, packed ahead of time.
__device__ __forceinline__ v16bf load_frag32B(const bf16_t* p) {
  union { uint4 q[2]; v16bf v; } U;
  U.q[0] = *(const uint4*)p;
  U.q[1] = *(const uint4*)(p + 8);
  return U.v;
}

// Load an A-fragment from an LDS row (16-bit A layout: lanes 0-15 take K
// {0..7,16..23}, lanes 16-31 take K {8..15,24..31} of the 32-wide k-slab).
__device__ __forceinline__ v16bf load_frag_a(const bf16_t* row, int kbase, int hi) {
  union { uint4 q[2]; v16bf v; } U;
  const bf16_t* p = row + kbase + hi * 8;
  U.q[0] = *(const uint4*)p;
  U.q[1] = *(const uint4*)(p + 16);
  return U.v;
}

// Pack W[G][K] (fp32, PyTorch gate-major) into bf16 WMMA-B fragment order:
// frag f = ktile*NT + ntile; element ((f*32)+lane)*16 + e.
__global__ void pack_w(const float* __restrict__ src, bf16_t* __restrict__ dst, int K, int G) {
  int idx = blockIdx.x * blockDim.x + threadIdx.x;
  if (idx >= K * G) return;
  int e    = idx & 15;
  int lane = (idx >> 4) & 31;
  int f    = idx >> 9;
  int NT   = G >> 4;
  int ktile = f / NT, ntile = f % NT;
  int nn = lane & 15, hiL = lane >> 4;
  int kk = (e < 8) ? (hiL * 8 + e) : (16 + hiL * 8 + (e - 8));
  int k = ktile * 32 + kk;
  int g = ntile * 16 + nn;
  dst[idx] = (bf16_t)src[g * K + k];
}

// x [B,T,IN] f32 -> xT [T,B,IN] bf16 (time-major for the scan; coalesced writes)
__global__ void x_to_bf16_tmajor(const float* __restrict__ x, bf16_t* __restrict__ xT) {
  size_t idx = (size_t)blockIdx.x * blockDim.x + threadIdx.x;
  int i = (int)(idx & (IN_N - 1));
  int b = (int)((idx >> 6) & (B_N - 1));
  int t = (int)(idx >> 15);
  xT[idx] = (bf16_t)x[((size_t)b * T_N + t) * IN_N + i];
}

// Fused GRU layer scan. One WG = 16 batch rows, 12 wave32s (waves 0-3: r,
// 4-7: z, 8-11: n with split x/h accumulators). All weights LDS-resident.
// Wave roles are forced uniform (readfirstlane) so WMMA runs with full EXEC
// under scalar branches (no saveexec around v_wmma).
template <int K_IN, bool LAST_ONLY>
__global__ __launch_bounds__(384, 1) void gru_scan(
    const bf16_t* __restrict__ xseq,      // [T, B, K_IN] bf16
    const bf16_t* __restrict__ wih_pack,  // packed [K_IN/32][24][32][16]
    const bf16_t* __restrict__ whh_pack,  // packed [4][24][32][16]
    const float*  __restrict__ b_ih,
    const float*  __restrict__ b_hh,
    bf16_t* __restrict__ outseq,          // [T,B,H] (if !LAST_ONLY)
    float*  __restrict__ outlast)         // [B,H]   (if LAST_ONLY)
{
  constexpr int SA  = K_IN + 8;   // padded x-tile row stride (bf16)
  constexpr int SH  = H_N + 8;    // padded h-tile row stride (bf16)
  constexpr int SRZ = H_N + 4;    // padded r/z row stride (f32)
  extern __shared__ char smem_raw[];
  bf16_t* wih_s = (bf16_t*)smem_raw;            // K_IN*G_N
  bf16_t* whh_s = wih_s + K_IN * G_N;           // H_N*G_N
  bf16_t* xT    = whh_s + H_N * G_N;            // 16*SA
  bf16_t* hT    = xT + 16 * SA;                 // 16*SH
  float*  Rb    = (float*)(hT + 16 * SH);       // 16*SRZ
  float*  Zb    = Rb + 16 * SRZ;                // 16*SRZ

  const int tid  = threadIdx.x;
  const int lane = tid & 31;
  // SGPR-uniform wave id -> scalar branches around WMMA blocks.
  const int wave = __builtin_amdgcn_readfirstlane(tid >> 5);
  const int b0   = blockIdx.x << 4;

  // Stage packed weights into LDS (one-time, ~150-220KB per WGP).
  {
    const uint4* s = (const uint4*)wih_pack;
    uint4* d = (uint4*)wih_s;
    for (int i = tid; i < K_IN * G_N / 8; i += 384) d[i] = s[i];
    const uint4* s2 = (const uint4*)whh_pack;
    uint4* d2 = (uint4*)whh_s;
    for (int i = tid; i < H_N * G_N / 8; i += 384) d2[i] = s2[i];
  }
  for (int i = tid; i < 16 * SH; i += 384) hT[i] = (bf16_t)0.f;  // h0 = 0
  __syncthreads();

  const int mlane = lane & 15;
  const int hi    = lane >> 4;
  const int nt0 = wave * 2, nt1 = nt0 + 1;           // two N-tiles per wave
  const int g0 = nt0 * 16 + mlane, g1 = nt1 * 16 + mlane;
  const bool isN = (wave >= 8);

  float biasA0, biasA1, biasB0 = 0.f, biasB1 = 0.f;
  if (!isN) {  // r/z: x-proj and h-proj fuse additively incl. both biases
    biasA0 = b_ih[g0] + b_hh[g0];
    biasA1 = b_ih[g1] + b_hh[g1];
  } else {     // n: keep x-part (b_ih) and h-part (b_hh) separate
    biasA0 = b_ih[g0]; biasB0 = b_hh[g0];
    biasA1 = b_ih[g1]; biasB1 = b_hh[g1];
  }

  for (int t = 0; t < T_N; ++t) {
    // Cooperative load of this step's 16 x K_IN input tile (contiguous in mem).
    const bf16_t* src = xseq + ((size_t)t * B_N + b0) * K_IN;
    for (int i = tid; i < 16 * K_IN / 2; i += 384) {
      int e = i * 2;
      int mm = e / K_IN, c = e - mm * K_IN;
      *(uint32_t*)(xT + mm * SA + c) = *(const uint32_t*)(src + e);
    }
    // Prefetch next step's tile into cache (global_prefetch_b8, speculative).
    if (wave == 0 && t + 1 < T_N) {
      const bf16_t* nsrc = xseq + ((size_t)(t + 1) * B_N + b0) * K_IN;
      if (lane * 64 < 16 * K_IN) __builtin_prefetch(nsrc + lane * 64, 0, 0);
    }
    __syncthreads();

    if (!isN) {
      // ---- r/z waves: one fused accumulator per tile ----
      v8f acc0, acc1;
#pragma unroll
      for (int i = 0; i < 8; ++i) { acc0[i] = biasA0; acc1[i] = biasA1; }
#pragma unroll
      for (int kt = 0; kt < K_IN / 32; ++kt) {
        v16bf a   = load_frag_a(xT + mlane * SA, kt * 32, hi);
        v16bf bb0 = load_frag32B(wih_s + ((size_t)((kt * NT_N) + nt0) * 32 + lane) * 16);
        v16bf bb1 = load_frag32B(wih_s + ((size_t)((kt * NT_N) + nt1) * 32 + lane) * 16);
        acc0 = wmma_bf16(a, bb0, acc0);
        acc1 = wmma_bf16(a, bb1, acc1);
      }
#pragma unroll
      for (int kt = 0; kt < H_N / 32; ++kt) {
        v16bf a   = load_frag_a(hT + mlane * SH, kt * 32, hi);
        v16bf bb0 = load_frag32B(whh_s + ((size_t)((kt * NT_N) + nt0) * 32 + lane) * 16);
        v16bf bb1 = load_frag32B(whh_s + ((size_t)((kt * NT_N) + nt1) * 32 + lane) * 16);
        acc0 = wmma_bf16(a, bb0, acc0);
        acc1 = wmma_bf16(a, bb1, acc1);
      }
      // publish sigmoid(r/z) to LDS
      float* dst = (wave < 4) ? Rb : Zb;
      const int j0 = g0 & (H_N - 1), j1 = g1 & (H_N - 1);
#pragma unroll
      for (int r = 0; r < 8; ++r) {
        int mr = r + hi * 8;  // C/D layout: vgpr r -> row r (lanes<16) / r+8
        dst[mr * SRZ + j0] = sigmoidf_(acc0[r]);
        dst[mr * SRZ + j1] = sigmoidf_(acc1[r]);
      }
      __syncthreads();  // R/Z visible
      __syncthreads();  // combine done
    } else {
      // ---- n waves: separate x-proj (accX) and h-proj (accH) ----
      v8f accX0, accX1, accH0, accH1;
#pragma unroll
      for (int i = 0; i < 8; ++i) {
        accX0[i] = biasA0; accX1[i] = biasA1; accH0[i] = biasB0; accH1[i] = biasB1;
      }
#pragma unroll
      for (int kt = 0; kt < K_IN / 32; ++kt) {
        v16bf a   = load_frag_a(xT + mlane * SA, kt * 32, hi);
        v16bf bb0 = load_frag32B(wih_s + ((size_t)((kt * NT_N) + nt0) * 32 + lane) * 16);
        v16bf bb1 = load_frag32B(wih_s + ((size_t)((kt * NT_N) + nt1) * 32 + lane) * 16);
        accX0 = wmma_bf16(a, bb0, accX0);
        accX1 = wmma_bf16(a, bb1, accX1);
      }
#pragma unroll
      for (int kt = 0; kt < H_N / 32; ++kt) {
        v16bf a   = load_frag_a(hT + mlane * SH, kt * 32, hi);
        v16bf bb0 = load_frag32B(whh_s + ((size_t)((kt * NT_N) + nt0) * 32 + lane) * 16);
        v16bf bb1 = load_frag32B(whh_s + ((size_t)((kt * NT_N) + nt1) * 32 + lane) * 16);
        accH0 = wmma_bf16(a, bb0, accH0);
        accH1 = wmma_bf16(a, bb1, accH1);
      }
      __syncthreads();  // wait for R/Z from r/z waves

      // n = tanh(xn + r*hn); h = (1-z)*n + z*h
      const int j0 = g0 - 256, j1 = g1 - 256;
#pragma unroll
      for (int r = 0; r < 8; ++r) {
        int mr = r + hi * 8;
        float rv0 = Rb[mr * SRZ + j0], zv0 = Zb[mr * SRZ + j0];
        float rv1 = Rb[mr * SRZ + j1], zv1 = Zb[mr * SRZ + j1];
        float nv0 = tanhf(accX0[r] + rv0 * accH0[r]);
        float nv1 = tanhf(accX1[r] + rv1 * accH1[r]);
        float hp0 = (float)hT[mr * SH + j0];
        float hp1 = (float)hT[mr * SH + j1];
        float hn0 = (1.f - zv0) * nv0 + zv0 * hp0;
        float hn1 = (1.f - zv1) * nv1 + zv1 * hp1;
        hT[mr * SH + j0] = (bf16_t)hn0;
        hT[mr * SH + j1] = (bf16_t)hn1;
      }
      __syncthreads();  // h-tile updated, visible to all waves
    }

    // Coalesced store of the updated 16x128 h tile (contiguous 4KB in [T,B,H]).
    if (!LAST_ONLY) {
      bf16_t* dstg = outseq + ((size_t)t * B_N + b0) * H_N;
      for (int i = tid; i < 16 * H_N / 2; i += 384) {
        int e = i * 2;
        int mm = e >> 7, c = e & (H_N - 1);
        *(uint32_t*)(dstg + e) = *(const uint32_t*)(hT + mm * SH + c);
      }
    }
  }

  if (LAST_ONLY) {
    for (int i = tid; i < 16 * H_N; i += 384) {
      int mm = i >> 7, c = i & (H_N - 1);
      outlast[(b0 + mm) * H_N + c] = (float)hT[mm * SH + c];
    }
  }
}

// y = relu(h2 @ fc1_w^T + fc1_b) @ fc2_w^T + fc2_b   (tiny: plain VALU)
__global__ __launch_bounds__(128) void fc_head(
    const float* __restrict__ h2, const float* __restrict__ w1,
    const float* __restrict__ bias1, const float* __restrict__ w2,
    const float* __restrict__ bias2, float* __restrict__ y)
{
  __shared__ float s[128];
  int b = blockIdx.x, j = threadIdx.x;
  const float* hr = h2 + b * H_N;
  float acc = bias1[j];
#pragma unroll 8
  for (int k = 0; k < H_N; ++k) acc = fmaf(hr[k], w1[j * H_N + k], acc);
  s[j] = fmaxf(acc, 0.f) * w2[j];
  __syncthreads();
  for (int off = 64; off > 0; off >>= 1) {
    if (j < off) s[j] += s[j + off];
    __syncthreads();
  }
  if (j == 0) y[b] = s[0] + bias2[0];
}

extern "C" void kernel_launch(void* const* d_in, const int* in_sizes, int n_in,
                              void* d_out, int out_size, void* d_ws, size_t ws_size,
                              hipStream_t stream) {
  (void)in_sizes; (void)n_in; (void)out_size; (void)ws_size;
  const float* x     = (const float*)d_in[0];
  const float* W_ih0 = (const float*)d_in[1];
  const float* W_hh0 = (const float*)d_in[2];
  const float* b_ih0 = (const float*)d_in[3];
  const float* b_hh0 = (const float*)d_in[4];
  const float* W_ih1 = (const float*)d_in[5];
  const float* W_hh1 = (const float*)d_in[6];
  const float* b_ih1 = (const float*)d_in[7];
  const float* b_hh1 = (const float*)d_in[8];
  const float* fc1_w = (const float*)d_in[9];
  const float* fc1_b = (const float*)d_in[10];
  const float* fc2_w = (const float*)d_in[11];
  const float* fc2_b = (const float*)d_in[12];
  float* y = (float*)d_out;

  char* ws = (char*)d_ws;
  size_t off = 0;
  bf16_t* xT = (bf16_t*)(ws + off); off += (size_t)T_N * B_N * IN_N * 2;  // 64 MB
  bf16_t* h1 = (bf16_t*)(ws + off); off += (size_t)T_N * B_N * H_N * 2;   // 128 MB
  float*  h2 = (float*)(ws + off);  off += (size_t)B_N * H_N * 4;
  bf16_t* pw_ih0 = (bf16_t*)(ws + off); off += (size_t)IN_N * G_N * 2;
  bf16_t* pw_hh0 = (bf16_t*)(ws + off); off += (size_t)H_N * G_N * 2;
  bf16_t* pw_ih1 = (bf16_t*)(ws + off); off += (size_t)H_N * G_N * 2;
  bf16_t* pw_hh1 = (bf16_t*)(ws + off); off += (size_t)H_N * G_N * 2;

  pack_w<<<(IN_N * G_N + 255) / 256, 256, 0, stream>>>(W_ih0, pw_ih0, IN_N, G_N);
  pack_w<<<(H_N * G_N + 255) / 256, 256, 0, stream>>>(W_hh0, pw_hh0, H_N, G_N);
  pack_w<<<(H_N * G_N + 255) / 256, 256, 0, stream>>>(W_ih1, pw_ih1, H_N, G_N);
  pack_w<<<(H_N * G_N + 255) / 256, 256, 0, stream>>>(W_hh1, pw_hh1, H_N, G_N);

  x_to_bf16_tmajor<<<(T_N * (size_t)B_N * IN_N) / 256, 256, 0, stream>>>(x, xT);

  size_t lds0 = (size_t)(IN_N * G_N + H_N * G_N + 16 * (IN_N + 8) + 16 * (H_N + 8)) * 2
              + 2 * 16 * (H_N + 4) * 4;  // ~171 KB
  gru_scan<IN_N, false><<<B_N / 16, 384, lds0, stream>>>(
      xT, pw_ih0, pw_hh0, b_ih0, b_hh0, h1, nullptr);

  size_t lds1 = (size_t)(H_N * G_N + H_N * G_N + 16 * (H_N + 8) + 16 * (H_N + 8)) * 2
              + 2 * 16 * (H_N + 4) * 4;  // ~222 KB (fits 320 KB WGP LDS)
  gru_scan<H_N, true><<<B_N / 16, 384, lds1, stream>>>(
      h1, pw_ih1, pw_hh1, b_ih1, b_hh1, nullptr, h2);

  fc_head<<<B_N, 128, 0, stream>>>(h2, fc1_w, fc1_b, fc2_w, fc2_b, y);
}
